// MHAttention_21706764714721
// MI455X (gfx1250) — compile-verified, matrix operand order checked by reading, not run on hardware
//
#include <hip/hip_runtime.h>
#include <hip/hip_bf16.h>

#define HEADS 4
#define DIN   128
#define DOUT  128
#define KA    256      // 2*DIN for attention MLP
#define SLOPE 0.01f

typedef __bf16 bf16;
typedef __attribute__((ext_vector_type(16))) __bf16 v16bf;
typedef __attribute__((ext_vector_type(8)))  float  v8f;

__device__ __forceinline__ float leaky(float x) { return x >= 0.f ? x : SLOPE * x; }

// order-preserving float -> uint key for atomicMax-based segment max
__device__ __forceinline__ unsigned fenc(float f) {
    unsigned u = __float_as_uint(f);
    return (u & 0x80000000u) ? ~u : (u | 0x80000000u);
}
__device__ __forceinline__ float fdec(unsigned k) {
    unsigned u = (k & 0x80000000u) ? (k & 0x7fffffffu) : ~k;
    return __uint_as_float(u);
}

// B fragment (32x16 bf16) from a row-major [out][K] weight matrix.
// lane q (=lane&15) covers output col o0+q; half=lane>>4 selects K 0..7/16..23 vs 8..15/24..31.
__device__ __forceinline__ v16bf load_bfrag(const bf16* __restrict__ Wrow, int b1) {
    v16bf v;
#pragma unroll
    for (int j = 0; j < 8; ++j) {
        v[j]     = Wrow[b1 + j];
        v[8 + j] = Wrow[b1 + 16 + j];
    }
    return v;
}

// ---------------- weight conversion f32 -> bf16 ----------------
__global__ void k_cvt(const float* __restrict__ m1, const float* __restrict__ m2,
                      const float* __restrict__ a1,
                      bf16* __restrict__ bm1, bf16* __restrict__ bm2, bf16* __restrict__ ba1) {
    int i = blockIdx.x * blockDim.x + threadIdx.x;
    if (i < HEADS * DOUT * DIN) { bm1[i] = (bf16)m1[i]; bm2[i] = (bf16)m2[i]; }
    if (i < HEADS * DIN * KA)   { ba1[i] = (bf16)a1[i]; }
}

// ---------------- init: zero out, reset segment stats ----------------
__global__ void k_init(float* __restrict__ out, unsigned* __restrict__ amax,
                       float* __restrict__ denom, int out_n, int seg_n) {
    int i = blockIdx.x * blockDim.x + threadIdx.x;
    if (i < out_n) out[i] = 0.f;
    if (i < seg_n) { amax[i] = 0u; denom[i] = 0.f; }
}

// ---------------- attention logits via WMMA ----------------
__global__ void __launch_bounds__(256)
k_alpha(const float* __restrict__ fea, const float* __restrict__ cry,
        const int* __restrict__ index,
        const bf16* __restrict__ wa1, const float* __restrict__ ba1,
        const float* __restrict__ wa2, const float* __restrict__ ba2,
        float* __restrict__ alpha, int N) {
    __shared__ bf16 sIn[8][16][KA];          // 64 KB: per-wave 16x256 bf16 A tiles
    const int wave = threadIdx.x >> 5;
    const int lane = threadIdx.x & 31;
    const long node0 = ((long)blockIdx.x * 8 + wave) * 16;
    const bool active = node0 < (long)N;
    const long nb = active ? node0 : (long)N - 16;

    {   // stage concat(fea, cry[index]) as bf16; 2 lanes per row
        const int r = lane >> 1, hs = lane & 1;
        const long n = nb + r;
        const float* src = hs ? (cry + (long)index[n] * DIN) : (fea + n * DIN);
        bf16* dst = &sIn[wave][r][hs * DIN];
#pragma unroll
        for (int c = 0; c < DIN; c += 4) {
            float4 v = *(const float4*)(src + c);
            dst[c] = (bf16)v.x; dst[c+1] = (bf16)v.y; dst[c+2] = (bf16)v.z; dst[c+3] = (bf16)v.w;
        }
    }
    __syncthreads();

    const int half = lane >> 4, q = lane & 15;
    v16bf af[8];                              // A fragments, K = 0..255 in 8 chunks of 32
#pragma unroll
    for (int kb = 0; kb < 8; ++kb) {
        const int b1 = kb * 32 + half * 8;
#pragma unroll
        for (int j = 0; j < 8; ++j) {
            af[kb][j]     = sIn[wave][q][b1 + j];
            af[kb][8 + j] = sIn[wave][q][b1 + 16 + j];
        }
    }

#pragma unroll
    for (int h = 0; h < HEADS; ++h) {
        float acc[8];
#pragma unroll
        for (int r = 0; r < 8; ++r) acc[r] = 0.f;
        const bf16* W = wa1 + (long)h * DIN * KA;
#pragma unroll
        for (int j = 0; j < 8; ++j) {
            const int o0 = j * 16;
            const float bias = ba1[h * DIN + o0 + q];
            v8f c;
#pragma unroll
            for (int r = 0; r < 8; ++r) c[r] = bias;
            const bf16* Wrow = W + (long)(o0 + q) * KA;
#pragma unroll
            for (int kb = 0; kb < 8; ++kb) {
                v16bf bf = load_bfrag(Wrow, kb * 32 + half * 8);
                c = __builtin_amdgcn_wmma_f32_16x16x32_bf16(false, af[kb], false, bf,
                                                            (short)0, c, false, false);
            }
            const float w2 = wa2[h * DIN + o0 + q];
#pragma unroll
            for (int r = 0; r < 8; ++r) acc[r] += leaky(c[r]) * w2;   // fold W_a2 dot
        }
        // reduce over the 16-lane column groups; row = r + 8*half
#pragma unroll
        for (int r = 0; r < 8; ++r) {
            float v = acc[r];
            v += __shfl_xor(v, 1, 32);
            v += __shfl_xor(v, 2, 32);
            v += __shfl_xor(v, 4, 32);
            v += __shfl_xor(v, 8, 32);
            if (active && q == 0)
                alpha[(node0 + r + 8 * half) * HEADS + h] = v + ba2[h];
        }
    }
}

// ---------------- segment max / exp-sum ----------------
__global__ void k_segmax(const float* __restrict__ alpha, const int* __restrict__ index,
                         unsigned* __restrict__ amax, int NH) {
    int i = blockIdx.x * blockDim.x + threadIdx.x;
    if (i >= NH) return;
    int n = i >> 2, h = i & 3;
    atomicMax(&amax[(long)index[n] * HEADS + h], fenc(alpha[i]));
}

__global__ void k_expsum(float* __restrict__ alpha, const int* __restrict__ index,
                         const unsigned* __restrict__ amax, float* __restrict__ denom, int NH) {
    int i = blockIdx.x * blockDim.x + threadIdx.x;
    if (i >= NH) return;
    int n = i >> 2, h = i & 3;
    long s = (long)index[n] * HEADS + h;
    float e = expf(alpha[i] - fdec(amax[s]));
    alpha[i] = e;
    atomicAdd(&denom[s], e);
}

// ---------------- message MLP via WMMA + weighted scatter-add ----------------
__global__ void __launch_bounds__(256)
k_main(const float* __restrict__ fea, const int* __restrict__ index,
       const bf16* __restrict__ wm1, const float* __restrict__ bm1,
       const bf16* __restrict__ wm2, const float* __restrict__ bm2,
       const float* __restrict__ ew, const float* __restrict__ denom,
       float* __restrict__ out, int N) {
    __shared__ bf16 sF[8][16][DIN];          // 32 KB fea tiles
    __shared__ bf16 sH[8][16][DIN];          // 32 KB hidden tiles
    const int wave = threadIdx.x >> 5;
    const int lane = threadIdx.x & 31;
    const long node0 = ((long)blockIdx.x * 8 + wave) * 16;
    const bool active = node0 < (long)N;
    const long nb = active ? node0 : (long)N - 16;

    {   // stage fea tile
        const int r = lane >> 1;
        const int cs = (lane & 1) * 64;
        const float* src = fea + (nb + r) * DIN + cs;
        bf16* dst = &sF[wave][r][cs];
#pragma unroll
        for (int c = 0; c < 64; c += 4) {
            float4 v = *(const float4*)(src + c);
            dst[c] = (bf16)v.x; dst[c+1] = (bf16)v.y; dst[c+2] = (bf16)v.z; dst[c+3] = (bf16)v.w;
        }
    }
    __syncthreads();

    const int half = lane >> 4, q = lane & 15;
    v16bf afF[4];
#pragma unroll
    for (int kb = 0; kb < 4; ++kb) {
        const int b1 = kb * 32 + half * 8;
#pragma unroll
        for (int j = 0; j < 8; ++j) {
            afF[kb][j]     = sF[wave][q][b1 + j];
            afF[kb][8 + j] = sF[wave][q][b1 + 16 + j];
        }
    }

    const long nlane = nb + q;               // lanes 0-15 / 16-31 mirror nodes 0..15
    const int  seg   = index[nlane];

    for (int h = 0; h < HEADS; ++h) {
        const float wl = ew[nlane * HEADS + h] / (denom[(long)seg * HEADS + h] + 1e-16f);
        const bf16* W1 = wm1 + (long)h * DOUT * DIN;
        // GEMM1: h1 = leaky(fea x W_m1^T + b) -> LDS as bf16
#pragma unroll
        for (int j = 0; j < 8; ++j) {
            const int o0 = j * 16;
            const float bias = bm1[h * DIN + o0 + q];
            v8f c;
#pragma unroll
            for (int r = 0; r < 8; ++r) c[r] = bias;
            const bf16* Wrow = W1 + (long)(o0 + q) * DIN;
#pragma unroll
            for (int kb = 0; kb < 4; ++kb)
                c = __builtin_amdgcn_wmma_f32_16x16x32_bf16(false, afF[kb], false,
                        load_bfrag(Wrow, kb * 32 + half * 8), (short)0, c, false, false);
#pragma unroll
            for (int r = 0; r < 8; ++r)
                sH[wave][r + 8 * half][o0 + q] = (bf16)leaky(c[r]);
        }
        __syncthreads();

        v16bf afH[4];
#pragma unroll
        for (int kb = 0; kb < 4; ++kb) {
            const int b1 = kb * 32 + half * 8;
#pragma unroll
            for (int j = 0; j < 8; ++j) {
                afH[kb][j]     = sH[wave][q][b1 + j];
                afH[kb][8 + j] = sH[wave][q][b1 + 16 + j];
            }
        }

        const bf16* W2 = wm2 + (long)h * DOUT * DIN;
        // GEMM2: m = h1 x W_m2^T + b, then weighted scatter-add
#pragma unroll
        for (int j = 0; j < 8; ++j) {
            const int o0 = j * 16;
            const float bias = bm2[h * DOUT + o0 + q];
            v8f c;
#pragma unroll
            for (int r = 0; r < 8; ++r) c[r] = bias;
            const bf16* Wrow = W2 + (long)(o0 + q) * DIN;
#pragma unroll
            for (int kb = 0; kb < 4; ++kb)
                c = __builtin_amdgcn_wmma_f32_16x16x32_bf16(false, afH[kb], false,
                        load_bfrag(Wrow, kb * 32 + half * 8), (short)0, c, false, false);
#pragma unroll
            for (int r = 0; r < 8; ++r) {
                const int   m  = r + 8 * half;           // node row in tile
                const float wr = __shfl(wl, m, 32);
                const int   sg = __shfl(seg, m, 32);
                if (active)
                    atomicAdd(&out[(long)sg * (HEADS * DOUT) + h * DOUT + o0 + q], c[r] * wr);
            }
        }
        __syncthreads();
    }
}

extern "C" void kernel_launch(void* const* d_in, const int* in_sizes, int n_in,
                              void* d_out, int out_size, void* d_ws, size_t ws_size,
                              hipStream_t stream) {
    (void)n_in; (void)out_size; (void)ws_size;
    const float* fea   = (const float*)d_in[0];
    const float* cry   = (const float*)d_in[1];
    const float* Wm1   = (const float*)d_in[2];
    const float* bm1   = (const float*)d_in[3];
    const float* Wm2   = (const float*)d_in[4];
    const float* bm2   = (const float*)d_in[5];
    const float* Wa1   = (const float*)d_in[6];
    const float* ba1   = (const float*)d_in[7];
    const float* Wa2   = (const float*)d_in[8];
    const float* ba2   = (const float*)d_in[9];
    const int*   index = (const int*)d_in[10];
    const int N = in_sizes[0] / DIN;
    const int S = in_sizes[1] / DIN;
    float* out = (float*)d_out;

    // workspace carve-out (all 256B aligned), ~4.5 MB total
    char* ws = (char*)d_ws;
    size_t off = 0;
    auto carve = [&](size_t bytes) {
        char* p = ws + off;
        off = (off + bytes + 255) & ~(size_t)255;
        return p;
    };
    float*    alpha = (float*)   carve((size_t)N * HEADS * sizeof(float));
    unsigned* amax  = (unsigned*)carve((size_t)S * HEADS * sizeof(unsigned));
    float*    denom = (float*)   carve((size_t)S * HEADS * sizeof(float));
    bf16* wbm1 = (bf16*)carve((size_t)HEADS * DOUT * DIN * sizeof(bf16));
    bf16* wbm2 = (bf16*)carve((size_t)HEADS * DOUT * DIN * sizeof(bf16));
    bf16* wba1 = (bf16*)carve((size_t)HEADS * DIN * KA   * sizeof(bf16));

    k_cvt<<<(HEADS * DIN * KA + 255) / 256, 256, 0, stream>>>(Wm1, Wm2, Wa1, wbm1, wbm2, wba1);

    const int out_n = S * HEADS * DOUT;
    k_init<<<(out_n + 255) / 256, 256, 0, stream>>>(out, amax, denom, out_n, S * HEADS);

    const int tiles  = (N + 15) / 16;
    const int blocks = (tiles + 7) / 8;
    k_alpha<<<blocks, 256, 0, stream>>>(fea, cry, index, wba1, ba1, Wa2, ba2, alpha, N);

    const int NH = N * HEADS;
    k_segmax<<<(NH + 255) / 256, 256, 0, stream>>>(alpha, index, amax, NH);
    k_expsum<<<(NH + 255) / 256, 256, 0, stream>>>(alpha, index, amax, denom, NH);

    k_main<<<blocks, 256, 0, stream>>>(fea, index, wbm1, bm1, wbm2, bm2, alpha, denom, out, N);
}
